// AttentionMechanism_84490596647559
// MI455X (gfx1250) — matrix operand order repaired and checked by reading, not compiled, not run on hardware
//
#include <hip/hip_runtime.h>
#include <stdint.h>

// Problem constants (reference: N=50000, H=8, M=D=128, fp32 in/out)
#define NROW 50000
#define HEADS 8
#define MDIM 128
#define DDIM 128
#define HM 1024           // H*M = per-row stride in elements
#define NB0 512           // sumsq(qs) partial blocks

typedef __attribute__((ext_vector_type(16))) __bf16 v16bf;
typedef __attribute__((ext_vector_type(8)))  __bf16 v8bf;
typedef __attribute__((ext_vector_type(8)))  float  v8f;

// Native bf16 convert (backend emits v_cvt; RNE) instead of manual bit-twiddling.
static __device__ __forceinline__ unsigned short f2bf(float f) {
  __bf16 b = (__bf16)f;
  return __builtin_bit_cast(unsigned short, b);
}
static __device__ __forceinline__ unsigned pack2bf(float lo, float hi) {
  return (unsigned)f2bf(lo) | ((unsigned)f2bf(hi) << 16);
}
static __device__ __forceinline__ float bf2f(unsigned short h) {
  return (float)__builtin_bit_cast(__bf16, h);
}

// ---------------------------------------------------------------- K0: sum(qs^2) partials
__global__ void __launch_bounds__(256) k0_sumsq(const float* __restrict__ q,
                                                float* __restrict__ part, int n4) {
  int t = blockIdx.x * 256 + threadIdx.x;
  int stride = gridDim.x * 256;
  const float4* q4 = (const float4*)q;
  float s = 0.f;
  for (int i = t; i < n4; i += stride) {
    float4 v = q4[i];
    s += v.x*v.x + v.y*v.y + v.z*v.z + v.w*v.w;
  }
  __shared__ float red[256];
  red[threadIdx.x] = s; __syncthreads();
  for (int o = 128; o > 0; o >>= 1) {
    if (threadIdx.x < o) red[threadIdx.x] += red[threadIdx.x + o];
    __syncthreads();
  }
  if (threadIdx.x == 0) part[blockIdx.x] = red[0];
}

// ---------------------------------------------------------------- K1: kvs partials + sums
// One WG per (chunk, head). C = Ks_chunk^T * Vs_chunk (128x128, f32 acc), K in steps of 32.
// Single-buffer LDS staging (no register prefetch -> no scratch spills); latency hidden by
// multiple resident WGs per WGP. Also computes partial sum(ks^2) and column sums of ks/vs.
__global__ void __launch_bounds__(256) k1_kvs(
    const float* __restrict__ ks, const float* __restrict__ vs,
    float* __restrict__ kvspart, float* __restrict__ ksspart,
    float* __restrict__ kcolpart, float* __restrict__ vcolpart,
    int chunkRows)
{
  const int c = blockIdx.x, h = blockIdx.y;
  const int t = threadIdx.x;
  const int wave = t >> 5, lane = t & 31;               // wave32
  const int base = c * chunkRows;
  const int rowEnd = min(NROW, base + chunkRows);
  const int steps = (rowEnd > base) ? ((rowEnd - base + 31) >> 5) : 0;

  // LDS: bf16 tiles, transposed [m][l] / [d][l], row pitch 40 halfs (80B: 16B-aligned, bank-spread)
  __shared__ unsigned short As[128 * 40];
  __shared__ unsigned short Bs[128 * 40];
  __shared__ float redc[128 * 16];
  __shared__ float red256[256];

  const int lp = (t >> 4) << 1;        // row pair within 32-row step: 0,2,..,30
  const int mq = (t & 15) * 8;         // 8-column slice: 0,8,..,120

  v8f acc[8] = {};
  float ksum_loc[8] = {}, vsum_loc[8] = {};
  float ss = 0.f;

  for (int s = 0; s < steps; s++) {
    // ---- stage: global f32 -> bf16 transposed LDS (+ streaming statistics)
    {
      int l0 = base + s * 32 + lp;
      const float4 z = make_float4(0.f, 0.f, 0.f, 0.f);
      float4 ka0, ka1, kb0, kb1, va0, va1, vb0, vb1;
      if (l0 < rowEnd) {
        const float4* p  = (const float4*)(ks + (size_t)l0 * HM + h * MDIM + mq);
        const float4* pv = (const float4*)(vs + (size_t)l0 * HM + h * DDIM + mq);
        ka0 = p[0];  ka1 = p[1];  va0 = pv[0]; va1 = pv[1];
      } else { ka0 = ka1 = va0 = va1 = z; }
      if (l0 + 1 < rowEnd) {
        const float4* p  = (const float4*)(ks + (size_t)(l0 + 1) * HM + h * MDIM + mq);
        const float4* pv = (const float4*)(vs + (size_t)(l0 + 1) * HM + h * DDIM + mq);
        kb0 = p[0];  kb1 = p[1];  vb0 = pv[0]; vb1 = pv[1];
      } else { kb0 = kb1 = vb0 = vb1 = z; }

      float ka[8] = {ka0.x, ka0.y, ka0.z, ka0.w, ka1.x, ka1.y, ka1.z, ka1.w};
      float kb[8] = {kb0.x, kb0.y, kb0.z, kb0.w, kb1.x, kb1.y, kb1.z, kb1.w};
      float va[8] = {va0.x, va0.y, va0.z, va0.w, va1.x, va1.y, va1.z, va1.w};
      float vb[8] = {vb0.x, vb0.y, vb0.z, vb0.w, vb1.x, vb1.y, vb1.z, vb1.w};
      #pragma unroll
      for (int i = 0; i < 8; i++) {
        *(unsigned*)&As[(mq + i) * 40 + lp] = pack2bf(ka[i], kb[i]);
        *(unsigned*)&Bs[(mq + i) * 40 + lp] = pack2bf(va[i], vb[i]);
        ksum_loc[i] += ka[i] + kb[i];
        vsum_loc[i] += va[i] + vb[i];
        ss += ka[i] * ka[i] + kb[i] * kb[i];
      }
    }
    __syncthreads();

    // ---- compute: 8 WMMAs per wave (one 16-row M stripe x all 8 D tiles)
    {
      // A frag (16x32 bf16): lanes<16 K0-7 & K16-23, lanes>=16 K8-15 & K24-31
      int arow = wave * 16 + (lane & 15);
      int akoff = (lane & 16) ? 8 : 0;
      v8bf alo = *(const v8bf*)&As[arow * 40 + akoff];
      v8bf ahi = *(const v8bf*)&As[arow * 40 + akoff + 16];
      v16bf af = __builtin_shufflevector(alo, ahi, 0,1,2,3,4,5,6,7,8,9,10,11,12,13,14,15);
      int boff = (lane & 16) ? 8 : 0;    // B frag: same K order as A frag
      #pragma unroll
      for (int dt = 0; dt < 8; dt++) {
        int brow = dt * 16 + (lane & 15);
        v8bf blo = *(const v8bf*)&Bs[brow * 40 + boff];
        v8bf bhi = *(const v8bf*)&Bs[brow * 40 + boff + 16];
        v16bf bf = __builtin_shufflevector(blo, bhi, 0,1,2,3,4,5,6,7,8,9,10,11,12,13,14,15);
        acc[dt] = __builtin_amdgcn_wmma_f32_16x16x32_bf16(
            false, af, false, bf, (short)0, acc[dt], false, false);
      }
    }
    __syncthreads();
  }

  // store 128x128 f32 partial
  float* outp = kvspart + ((size_t)(c * HEADS + h) << 14);
  #pragma unroll
  for (int dt = 0; dt < 8; dt++) {
    int d = dt * 16 + (lane & 15);
    #pragma unroll
    for (int j = 0; j < 8; j++) {
      int m = wave * 16 + j + ((lane >> 4) << 3);
      outp[m * 128 + d] = acc[dt][j];
    }
  }

  // sum(ks^2) partial
  red256[t] = ss; __syncthreads();
  for (int o = 128; o > 0; o >>= 1) {
    if (t < o) red256[t] += red256[t + o];
    __syncthreads();
  }
  if (t == 0) ksspart[c * HEADS + h] = red256[0];

  // column-sum partials (ks then vs), 16 partials per column
  #pragma unroll
  for (int i = 0; i < 8; i++) redc[(mq + i) * 16 + (t >> 4)] = ksum_loc[i];
  __syncthreads();
  if (t < 128) {
    float s2 = 0.f;
    #pragma unroll
    for (int i = 0; i < 16; i++) s2 += redc[t * 16 + i];
    kcolpart[(c * HEADS + h) * 128 + t] = s2;
  }
  __syncthreads();
  #pragma unroll
  for (int i = 0; i < 8; i++) redc[(mq + i) * 16 + (t >> 4)] = vsum_loc[i];
  __syncthreads();
  if (t < 128) {
    float s2 = 0.f;
    #pragma unroll
    for (int i = 0; i < 16; i++) s2 += redc[t * 16 + i];
    vcolpart[(c * HEADS + h) * 128 + t] = s2;
  }
}

// ---------------------------------------------------------------- K2a: norms -> scale; reduce col sums
__global__ void __launch_bounds__(256) k2a_finalize(
    const float* __restrict__ qss, const float* __restrict__ kss,
    const float* __restrict__ kcol, const float* __restrict__ vcol,
    float* __restrict__ scaleOut, float* __restrict__ kssum,
    float* __restrict__ vssum, int split)
{
  __shared__ float red[256];
  __shared__ float sScale;
  int t = threadIdx.x;
  float s = qss[t] + qss[t + 256];
  red[t] = s; __syncthreads();
  for (int o = 128; o > 0; o >>= 1) { if (t < o) red[t] += red[t + o]; __syncthreads(); }
  float sq = red[0];
  __syncthreads();
  int nk = split * HEADS;
  float s2 = 0.f;
  for (int i = t; i < nk; i += 256) s2 += kss[i];
  red[t] = s2; __syncthreads();
  for (int o = 128; o > 0; o >>= 1) { if (t < o) red[t] += red[t + o]; __syncthreads(); }
  if (t == 0) {
    float sc = rsqrtf(red[0] * sq);    // 1/(||q|| * ||k||)
    sScale = sc; scaleOut[0] = sc;
  }
  __syncthreads();
  float sc = sScale;
  for (int o = t; o < HEADS * MDIM; o += 256) {
    float a = 0.f, b = 0.f;
    for (int cI = 0; cI < split; cI++) { a += kcol[cI * 1024 + o]; b += vcol[cI * 1024 + o]; }
    kssum[o] = a * sc;   // pre-scaled ks_sum
    vssum[o] = b;        // raw vs_sum
  }
}

// ---------------------------------------------------------------- K2b: reduce kvs partials -> scaled bf16 kvs^T
__global__ void __launch_bounds__(256) k2b_kvs_reduce(
    const float* __restrict__ kvspart, const float* __restrict__ scale,
    unsigned short* __restrict__ kvsT, int split)
{
  int e = blockIdx.x * 256 + threadIdx.x;      // [0, 8*16384)
  int h = e >> 14;
  int r = e & 16383;                           // m*128 + d
  int m = r >> 7, d = r & 127;
  float s = 0.f;
  for (int cI = 0; cI < split; cI++) s += kvspart[((size_t)(cI * HEADS + h) << 14) + r];
  kvsT[((size_t)h << 14) + d * 128 + m] = f2bf(s * scale[0]);   // transposed [d][m]
}

// ---------------------------------------------------------------- K3: out = (Qs·kvs + vs_sum) / (Qs·ks_sum + N)
__global__ void __launch_bounds__(256) k3_out(
    const float* __restrict__ qs, const unsigned short* __restrict__ kvsT,
    const float* __restrict__ kssum, const float* __restrict__ vssum,
    float* __restrict__ out)
{
  const int h = blockIdx.y;
  const int n0 = blockIdx.x * 128;
  const int t = threadIdx.x, wave = t >> 5, lane = t & 31;
  __shared__ unsigned short Qs[128 * 136];   // [n][m], pitch 136 halfs (272B)
  __shared__ unsigned short KV[128 * 136];   // [d][m]
  __shared__ float kss_l[128], vss_l[128], dinv[128];

  { // stage kvs^T (bf16, no conversion needed) with padded pitch
    const uint4* src = (const uint4*)(kvsT + ((size_t)h << 14));
    #pragma unroll
    for (int k = 0; k < 8; k++) {
      int g8 = k * 256 + t;
      int g = g8 * 8;
      int d = g >> 7, off = g & 127;
      *(uint4*)&KV[d * 136 + off] = src[g8];
    }
  }
  if (t < 128) { kss_l[t] = kssum[h * 128 + t]; vss_l[t] = vssum[h * 128 + t]; }
  { // stage qs tile as bf16 (zero-pad ragged tail)
    int mq = (t & 15) * 8;
    int rb = (t >> 4) << 1;
    #pragma unroll
    for (int qd = 0; qd < 4; qd++) {
      #pragma unroll
      for (int rr = 0; rr < 2; rr++) {
        int r = rb + rr + qd * 32;
        int n = n0 + r;
        float4 a, b;
        if (n < NROW) {
          const float4* p = (const float4*)(qs + (size_t)n * HM + h * MDIM + mq);
          a = p[0]; b = p[1];
        } else { a = b = make_float4(0.f, 0.f, 0.f, 0.f); }
        uint4 pk;
        pk.x = pack2bf(a.x, a.y);
        pk.y = pack2bf(a.z, a.w);
        pk.z = pack2bf(b.x, b.y);
        pk.w = pack2bf(b.z, b.w);
        *(uint4*)&Qs[r * 136 + mq] = pk;
      }
    }
  }
  __syncthreads();

  // per-row normalizer (kssum is pre-scaled; +N)
  if (t < 128) {
    float dot = 0.f;
    for (int m = 0; m < 128; m++) dot += bf2f(Qs[t * 136 + m]) * kss_l[m];
    dinv[t] = 1.f / (dot + (float)NROW);
  }

  v8f acc[8] = {};
  #pragma unroll
  for (int kk = 0; kk < 4; kk++) {
    int arow = wave * 16 + (lane & 15);
    int akoff = kk * 32 + ((lane & 16) ? 8 : 0);
    v8bf alo = *(const v8bf*)&Qs[arow * 136 + akoff];
    v8bf ahi = *(const v8bf*)&Qs[arow * 136 + akoff + 16];
    v16bf af = __builtin_shufflevector(alo, ahi, 0,1,2,3,4,5,6,7,8,9,10,11,12,13,14,15);
    int boff = kk * 32 + ((lane & 16) ? 8 : 0);
    #pragma unroll
    for (int dt = 0; dt < 8; dt++) {
      int brow = dt * 16 + (lane & 15);
      v8bf blo = *(const v8bf*)&KV[brow * 136 + boff];
      v8bf bhi = *(const v8bf*)&KV[brow * 136 + boff + 16];
      v16bf bf = __builtin_shufflevector(blo, bhi, 0,1,2,3,4,5,6,7,8,9,10,11,12,13,14,15);
      acc[dt] = __builtin_amdgcn_wmma_f32_16x16x32_bf16(
          false, af, false, bf, (short)0, acc[dt], false, false);
    }
  }
  __syncthreads();

  #pragma unroll
  for (int dt = 0; dt < 8; dt++) {
    int col = dt * 16 + (lane & 15);
    float vsc = vss_l[col];
    #pragma unroll
    for (int j = 0; j < 8; j++) {
      int row = wave * 16 + j + ((lane >> 4) << 3);
      int n = n0 + row;
      if (n < NROW)
        out[(size_t)n * HM + h * DDIM + col] = (acc[dt][j] + vsc) * dinv[row];
    }
  }
}

// ---------------------------------------------------------------- host launcher
extern "C" void kernel_launch(void* const* d_in, const int* in_sizes, int n_in,
                              void* d_out, int out_size, void* d_ws, size_t ws_size,
                              hipStream_t stream) {
  const float* qs = (const float*)d_in[0];
  const float* ks = (const float*)d_in[1];
  const float* vs = (const float*)d_in[2];
  float* out = (float*)d_out;
  float* ws = (float*)d_ws;

  // pick split so kvs partials fit in workspace
  int split = 32;
  for (;;) {
    size_t need = 4ull * ((size_t)split * 131072 + 512 + 256 +
                          (size_t)split * 2048 + 4 + 2048 + 65536);
    if (need <= ws_size || split == 1) break;
    split >>= 1;
  }
  int chunkRows = (NROW + split - 1) / split;

  size_t o = 0;
  float* kvspart = ws + o;  o += (size_t)split * 131072;  // split*8*128*128 f32
  float* qss     = ws + o;  o += 512;                     // sumsq(qs) partials
  float* kss     = ws + o;  o += 256;                     // sumsq(ks) partials
  float* kcol    = ws + o;  o += (size_t)split * 1024;    // ks col-sum partials
  float* vcol    = ws + o;  o += (size_t)split * 1024;    // vs col-sum partials
  float* scale   = ws + o;  o += 4;
  float* kssum   = ws + o;  o += 1024;                    // scaled ks_sum [h][m]
  float* vssum   = ws + o;  o += 1024;                    // vs_sum [h][d]
  unsigned short* kvsT = (unsigned short*)(ws + o);       // bf16 kvs^T [h][d][m]

  k0_sumsq<<<NB0, 256, 0, stream>>>(qs, qss, (NROW * HM) / 4);

  dim3 g1(split, HEADS);
  k1_kvs<<<g1, 256, 0, stream>>>(ks, vs, kvspart, kss, kcol, vcol, chunkRows);

  k2a_finalize<<<1, 256, 0, stream>>>(qss, kss, kcol, vcol, scale, kssum, vssum, split);

  k2b_kvs_reduce<<<512, 256, 0, stream>>>(kvspart, scale, kvsT, split);

  dim3 g3((NROW + 127) / 128, HEADS);
  k3_out<<<g3, 256, 0, stream>>>(qs, kvsT, kssum, vssum, out);
}
